// HyperAttention_89464168775953
// MI455X (gfx1250) — matrix operand order repaired and checked by reading, not compiled, not run on hardware
//
#include <hip/hip_runtime.h>

// HyperAttention for MI455X (gfx1250), wave32 + bf16 WMMA + async global->LDS.
// B=2 H=16 N=8192 D=64, P=8 (256 buckets), BLOCK=256, SAMPLE=256.

#define Bdim   2
#define Hdim   16
#define Ndim   8192
#define Ddim   64
#define BHdim  (Bdim*Hdim)          // 32
#define BHN    (BHdim*Ndim)         // 262144
#define QROWS  128                  // query rows per attention workgroup
#define SMP    256

// ---- LDS layout of the fused attention kernel (byte offsets) ----
#define OFF_Q    0u                 // 128*64 bf16   = 16384
#define OFF_KB   16384u             // 256*64 bf16   = 32768
#define OFF_VB   49152u             // 256*64 bf16   = 32768
#define OFF_S    81920u             // 128*256 f32   = 131072
#define OFF_KM   212992u            // 256*64 bf16   = 32768
#define OFF_VM   245760u            // 256*64 bf16   = 32768
#define OFF_BIAS 278528u            // 256 f32       = 1024
#define OFF_LSE  279552u            // 128 f32       = 512
#define OFF_C    280064u            // 128 f32       = 512
#define OFF_SIDX 280576u            // 256 i32       = 1024
#define OFF_OROW 281600u            // 128 i32       = 512
#define SMEM_BYTES 282112u          // ~275.5 KB < 320 KB/WGP

typedef __attribute__((ext_vector_type(16))) __bf16          v16bf;
typedef __attribute__((ext_vector_type(16))) unsigned short  v16us;
typedef __attribute__((ext_vector_type(8)))  float           v8f;

union BfFrag { v16us us; unsigned int u32[8]; v16bf bf; };

__device__ __forceinline__ unsigned short f2bf(float x) {
  unsigned int u = __float_as_uint(x);
  u += 0x7FFFu + ((u >> 16) & 1u);           // round-to-nearest-even
  return (unsigned short)(u >> 16);
}

// 16-byte global -> LDS async copy (GLOBAL_LOAD_ASYNC_TO_LDS_B128, ASYNCcnt)
__device__ __forceinline__ void async_copy16(unsigned lds_off, const void* g) {
  unsigned long long ga = (unsigned long long)(size_t)g;
  asm volatile("global_load_async_to_lds_b128 %0, %1, off"
               :: "v"(lds_off), "v"(ga) : "memory");
}
__device__ __forceinline__ void async_wait0() {
  asm volatile("s_wait_asynccnt 0x0" ::: "memory");
}

// ---- WMMA fragment loaders (bf16 16x16x32, wave32; contiguous k = 16h + s in both A and B) ----

// A (16x32, MxK) from row-major ushort src[(m0+m)*ld + k0 + k]
__device__ __forceinline__ v16bf load_frag_A(const unsigned short* src, int ld,
                                             int m0, int k0) {
  int lane = threadIdx.x & 31, m = lane & 15, h = lane >> 4;
  const unsigned short* row = src + (size_t)(m0 + m) * ld + k0;
  BfFrag f;
#pragma unroll
  for (int j = 0; j < 8; ++j) {
    int kb = 2*j + 16*h;
    f.u32[j] = *(const unsigned int*)(row + kb);
  }
  return f.bf;
}

// A from f32 source (converted to bf16 on the fly)
__device__ __forceinline__ v16bf load_frag_A_f32(const float* src, int ld,
                                                 int m0, int k0) {
  int lane = threadIdx.x & 31, m = lane & 15, h = lane >> 4;
  const float* row = src + (size_t)(m0 + m) * ld + k0;
  BfFrag f;
#pragma unroll
  for (int j = 0; j < 8; ++j) {
    int kb = 2*j + 16*h;
    f.us[2*j]   = f2bf(row[kb]);
    f.us[2*j+1] = f2bf(row[kb + 1]);
  }
  return f.bf;
}

// B (32x16, KxN) where B[k][n] = src[(n0+n)*ld + k0 + k]  (transposed source, e.g. K^T)
__device__ __forceinline__ v16bf load_frag_BT(const unsigned short* src, int ld,
                                              int n0, int k0) {
  int lane = threadIdx.x & 31, n = lane & 15, h = lane >> 4;
  const unsigned short* row = src + (size_t)(n0 + n) * ld + k0;
  BfFrag f;
#pragma unroll
  for (int j = 0; j < 8; ++j) {
    int kb = 2*j + 16*h;
    f.u32[j] = *(const unsigned int*)(row + kb);
  }
  return f.bf;
}

// B (32x16, KxN) where B[k][n] = src[(k0+k)*ld + n0 + n]  (natural row-major, e.g. V)
__device__ __forceinline__ v16bf load_frag_BN(const unsigned short* src, int ld,
                                              int k0, int n0) {
  int lane = threadIdx.x & 31, n = lane & 15, h = lane >> 4;
  BfFrag f;
#pragma unroll
  for (int j = 0; j < 8; ++j) {
    int k = k0 + 2*j + 16*h;
    f.us[2*j]   = src[(size_t)k * ld + n0 + n];
    f.us[2*j+1] = src[(size_t)(k + 1) * ld + n0 + n];
  }
  return f.bf;
}

// ---- kernel 1: LSH hash: code = Gray(bits of x . proj) ----
__global__ __launch_bounds__(256) void hash_kernel(const float* __restrict__ x,
                                                   const float* __restrict__ proj,
                                                   int* __restrict__ code) {
  __shared__ float pr[64 * 8];
  int t = threadIdx.x;
  for (int i = t; i < 512; i += 256) pr[i] = proj[i];
  __syncthreads();
  size_t row = (size_t)blockIdx.x * 256 + t;
  if (row >= (size_t)BHN) return;
  const float* xr = x + row * Ddim;
  float acc[8] = {0.f,0.f,0.f,0.f,0.f,0.f,0.f,0.f};
#pragma unroll 4
  for (int d = 0; d < Ddim; ++d) {
    float xv = xr[d];
#pragma unroll
    for (int p = 0; p < 8; ++p) acc[p] += xv * pr[d * 8 + p];
  }
  int c = 0;
#pragma unroll
  for (int p = 0; p < 8; ++p) c |= (acc[p] > 0.f) ? (1 << p) : 0;
  code[row] = c ^ (c >> 1);                 // PERM == binary-reflected Gray code
}

// ---- kernel 2: stable counting sort of 8192 8-bit codes per (b,h) ----
__global__ __launch_bounds__(256) void sort_kernel(const int* __restrict__ code,
                                                   int* __restrict__ idx) {
  __shared__ unsigned char codes[Ndim];
  __shared__ int hist[256];
  __shared__ int offs[256];
  int bh = blockIdx.x, t = threadIdx.x;
  const int* cg = code + (size_t)bh * Ndim;
  hist[t] = 0;
  for (int i = t; i < Ndim; i += 256) codes[i] = (unsigned char)cg[i];
  __syncthreads();
  for (int i = t; i < Ndim; i += 256) atomicAdd(&hist[codes[i]], 1);
  __syncthreads();
  if (t == 0) { int s = 0; for (int b = 0; b < 256; ++b) { offs[b] = s; s += hist[b]; } }
  __syncthreads();
  int o = offs[t];                           // thread t owns bucket t -> stable
  int* og = idx + (size_t)bh * Ndim;
  for (int i = 0; i < Ndim; ++i)
    if (codes[i] == (unsigned char)t) og[o++] = i;
}

// ---- kernel 3: gather sorted rows into contiguous bf16 (Q pre-scaled) ----
__global__ __launch_bounds__(256) void gather_kernel(const float* __restrict__ q,
                                                     const float* __restrict__ k,
                                                     const float* __restrict__ v,
                                                     const int* __restrict__ q_idx,
                                                     const int* __restrict__ k_idx,
                                                     unsigned short* __restrict__ q_s,
                                                     unsigned short* __restrict__ k_s,
                                                     unsigned short* __restrict__ v_s,
                                                     float scale) {
  size_t gid = (size_t)blockIdx.x * 256 + threadIdx.x;   // over BHN*64
  int col = (int)(gid & 63);
  size_t j = gid >> 6;
  size_t bh = j >> 13;
  size_t base = bh << 13;
  int qi = q_idx[j], ki = k_idx[j];
  q_s[gid] = f2bf(q[(base + qi) * Ddim + col] * scale);
  k_s[gid] = f2bf(k[(base + ki) * Ddim + col]);
  v_s[gid] = f2bf(v[(base + ki) * Ddim + col]);
}

// ---- kernel 4 (fused): block-diag attention + sampled residual + merge ----
__global__ __launch_bounds__(256) void attn_kernel(const unsigned short* __restrict__ q_s,
                                                   const unsigned short* __restrict__ k_s,
                                                   const unsigned short* __restrict__ v_s,
                                                   const int* __restrict__ sample_idx,
                                                   const int* __restrict__ q_idx,
                                                   float* __restrict__ out) {
  extern __shared__ char smem[];
  unsigned short* Qs   = (unsigned short*)(smem + OFF_Q);
  unsigned short* Kb   = (unsigned short*)(smem + OFF_KB);
  unsigned short* Vb   = (unsigned short*)(smem + OFF_VB);
  float*          Ss   = (float*)(smem + OFF_S);
  unsigned short* Km   = (unsigned short*)(smem + OFF_KM);
  unsigned short* Vm   = (unsigned short*)(smem + OFF_VM);
  float*          bias = (float*)(smem + OFF_BIAS);
  float*          lseb = (float*)(smem + OFF_LSE);
  float*          cbuf = (float*)(smem + OFF_C);
  int*            sidx = (int*)(smem + OFF_SIDX);
  int*            orow = (int*)(smem + OFF_OROW);

  int wg = blockIdx.x;
  int nstrips = Ndim / QROWS;                // 64
  int strip = wg % nstrips, bh = wg / nstrips;
  int j0 = strip * QROWS;
  int kr = (j0 >> 8) << 8;                   // key block base row (BLOCK=256)
  int qblk = j0 >> 8;
  int t = threadIdx.x;

  sidx[t] = sample_idx[(size_t)bh * SMP + t];

  // async global -> LDS fills (16 B per lane per issue; ASYNCcnt tracked)
  const char* qg = (const char*)(q_s + ((size_t)bh * Ndim + j0) * Ddim);
  const char* kg = (const char*)(k_s + ((size_t)bh * Ndim + kr) * Ddim);
  const char* vg = (const char*)(v_s + ((size_t)bh * Ndim + kr) * Ddim);
  for (int c = t; c < 1024; c += 256) async_copy16(OFF_Q  + c * 16, qg + c * 16);
  for (int c = t; c < 2048; c += 256) async_copy16(OFF_KB + c * 16, kg + c * 16);
  for (int c = t; c < 2048; c += 256) async_copy16(OFF_VB + c * 16, vg + c * 16);
  __syncthreads();                            // sidx visible to all threads

  bias[t] = ((sidx[t] >> 8) == qblk) ? -3.402823466e38f : 0.f;
  if (t < QROWS) orow[t] = q_idx[(size_t)bh * Ndim + j0 + t];
  size_t kbase = (size_t)bh * Ndim;
  for (int c = t; c < 2048; c += 256) {       // 256 sampled rows x 8 chunks
    int s = c >> 3, ch = (c & 7) * 16;
    const char* krow = (const char*)(k_s + (kbase + sidx[s]) * Ddim);
    const char* vrow = (const char*)(v_s + (kbase + sidx[s]) * Ddim);
    async_copy16(OFF_KM + c * 16, krow + ch);
    async_copy16(OFF_VM + c * 16, vrow + ch);
  }
  async_wait0();
  __syncthreads();

  int wave = t >> 5, lane = t & 31;
  int m0 = wave * 16;
  int n = lane & 15, h = lane >> 4;

  // ---- stage 1 (block-diag): S = Q Kb^T  (scale folded into Q) ----
  for (int nt = 0; nt < 16; ++nt) {
    v8f acc = {};
#pragma unroll
    for (int ks = 0; ks < 2; ++ks) {
      v16bf a = load_frag_A (Qs, Ddim, m0, ks * 32);
      v16bf b = load_frag_BT(Kb, Ddim, nt * 16, ks * 32);
      acc = __builtin_amdgcn_wmma_f32_16x16x32_bf16(false, a, false, b,
                                                    (short)0, acc, false, false);
    }
#pragma unroll
    for (int r = 0; r < 8; ++r)
      Ss[(size_t)(m0 + r + 8*h) * 256 + nt * 16 + n] = acc[r];
  }
  __syncthreads();

  // ---- softmax (block): p = exp(s - lse) in place; lse -> LDS ----
  if (t < QROWS) {
    float* row = Ss + (size_t)t * 256;
    float m = -3.402823466e38f;
    for (int i = 0; i < 256; ++i) m = fmaxf(m, row[i]);
    float l = 0.f;
    for (int i = 0; i < 256; ++i) l += __expf(row[i] - m);
    float lse = m + __logf(l);
    for (int i = 0; i < 256; ++i) row[i] = __expf(row[i] - lse);
    lseb[t] = lse;
  }
  __syncthreads();

  // ---- stage 2 (block): O_blk = P Vb, kept in registers ----
  v8f oblk[4];
  for (int nt = 0; nt < 4; ++nt) {
    v8f acc = {};
#pragma unroll
    for (int ks = 0; ks < 8; ++ks) {
      v16bf a = load_frag_A_f32(Ss, 256, m0, ks * 32);
      v16bf b = load_frag_BN(Vb, Ddim, ks * 32, nt * 16);
      acc = __builtin_amdgcn_wmma_f32_16x16x32_bf16(false, a, false, b,
                                                    (short)0, acc, false, false);
    }
    oblk[nt] = acc;
  }
  __syncthreads();

  // ---- stage 1 (residual): S = Q Km^T ----
  for (int nt = 0; nt < 16; ++nt) {
    v8f acc = {};
#pragma unroll
    for (int ks = 0; ks < 2; ++ks) {
      v16bf a = load_frag_A (Qs, Ddim, m0, ks * 32);
      v16bf b = load_frag_BT(Km, Ddim, nt * 16, ks * 32);
      acc = __builtin_amdgcn_wmma_f32_16x16x32_bf16(false, a, false, b,
                                                    (short)0, acc, false, false);
    }
#pragma unroll
    for (int r = 0; r < 8; ++r)
      Ss[(size_t)(m0 + r + 8*h) * 256 + nt * 16 + n] = acc[r];
  }
  __syncthreads();

  // ---- softmax (residual, with overlap bias); c = sigmoid(lse_blk - lse_res) ----
  if (t < QROWS) {
    float* row = Ss + (size_t)t * 256;
    float m = -3.402823466e38f;
    for (int i = 0; i < 256; ++i) { float v = row[i] + bias[i]; row[i] = v; m = fmaxf(m, v); }
    float l = 0.f;
    for (int i = 0; i < 256; ++i) l += __expf(row[i] - m);
    float lse = m + __logf(l);
    for (int i = 0; i < 256; ++i) row[i] = __expf(row[i] - lse);
    float lser = lse + 3.4657359028f;         // + log(N/SAMPLE) = log 32
    cbuf[t] = 1.f / (1.f + __expf(lser - lseb[t]));
  }
  __syncthreads();

  // ---- stage 2 (residual) + merge + unsort-scatter ----
  for (int nt = 0; nt < 4; ++nt) {
    v8f acc = {};
#pragma unroll
    for (int ks = 0; ks < 8; ++ks) {
      v16bf a = load_frag_A_f32(Ss, 256, m0, ks * 32);
      v16bf b = load_frag_BN(Vm, Ddim, ks * 32, nt * 16);
      acc = __builtin_amdgcn_wmma_f32_16x16x32_bf16(false, a, false, b,
                                                    (short)0, acc, false, false);
    }
#pragma unroll
    for (int r = 0; r < 8; ++r) {
      int row = m0 + r + 8*h;
      float c = cbuf[row];
      float val = c * oblk[nt][r] + (1.f - c) * acc[r];
      out[((size_t)bh * Ndim + orow[row]) * Ddim + nt * 16 + n] = val;
    }
  }
}

extern "C" void kernel_launch(void* const* d_in, const int* in_sizes, int n_in,
                              void* d_out, int out_size, void* d_ws, size_t ws_size,
                              hipStream_t stream) {
  const float* q    = (const float*)d_in[0];
  const float* k    = (const float*)d_in[1];
  const float* v    = (const float*)d_in[2];
  const float* proj = (const float*)d_in[3];
  const int*   smp  = (const int*)d_in[4];
  float* out = (float*)d_out;

  char* w = (char*)d_ws;
  const size_t MB = 1024ull * 1024ull;
  int*            qcode = (int*)(w + 0 * MB);              // 1 MB
  int*            kcode = (int*)(w + 1 * MB);              // 1 MB
  int*            q_idx = (int*)(w + 2 * MB);              // 1 MB
  int*            k_idx = (int*)(w + 3 * MB);              // 1 MB
  unsigned short* q_s   = (unsigned short*)(w + 4 * MB);   // 32 MB
  unsigned short* k_s   = (unsigned short*)(w + 36 * MB);  // 32 MB
  unsigned short* v_s   = (unsigned short*)(w + 68 * MB);  // 32 MB (100 MB total)

  const float scale = 0.125f;                              // 64^-0.5

  hash_kernel<<<BHN / 256, 256, 0, stream>>>(q, proj, qcode);
  hash_kernel<<<BHN / 256, 256, 0, stream>>>(k, proj, kcode);
  sort_kernel<<<BHdim, 256, 0, stream>>>(qcode, q_idx);
  sort_kernel<<<BHdim, 256, 0, stream>>>(kcode, k_idx);
  gather_kernel<<<(BHN * Ddim) / 256, 256, 0, stream>>>(q, k, v, q_idx, k_idx,
                                                        q_s, k_s, v_s, scale);

  int attn_wgs = BHdim * (Ndim / QROWS);                   // 2048
  attn_kernel<<<attn_wgs, 256, SMEM_BYTES, stream>>>(q_s, k_s, v_s, smp, q_idx, out);
}
